// VQCodebook_1039382086317
// MI455X (gfx1250) — matrix operand order repaired and checked by reading, not compiled, not run on hardware
//
#include <hip/hip_runtime.h>

typedef _Float16 h8   __attribute__((ext_vector_type(8)));
typedef _Float16 v16h __attribute__((ext_vector_type(16)));
typedef float    v8f  __attribute__((ext_vector_type(8)));

#define BATCH  16
#define EMB    64
#define CODES  16384
#define NTOK   1024

#define QTILE  128          // queries per block
#define TCHUNK 512          // tokens staged per LDS chunk
#define ROWP   72           // padded LDS row stride (halves): 36 dwords -> conflict-free
#define NCHAIN 4            // independent accumulator chains per loop iteration
#define BIAS   128.0f       // keeps score = dot + BIAS - 0.5*e2 strictly positive (>12 sigma)

// dynamic LDS layout (bytes)
#define X_OFF      0
#define X_BYTES    (QTILE * ROWP * 2)        // 18432
#define C_OFF      (X_OFF + X_BYTES)
#define C_BYTES    (TCHUNK * ROWP * 2)       // 73728
#define E_OFF      (C_OFF + C_BYTES)
#define E_BYTES    (TCHUNK * 4)              // 2048
#define SMEM_BYTES (E_OFF + E_BYTES)         // 94208 < 320KB WGP LDS

__device__ __forceinline__ v16h load_bfrag(const _Float16* p) {
    h8 lo = *(const h8*)(p);
    h8 hi = *(const h8*)(p + 16);
    v16h r;
#pragma unroll
    for (int i = 0; i < 8; ++i) { r[i] = lo[i]; r[8 + i] = hi[i]; }
    return r;
}

__global__ __launch_bounds__(256) void vq_argmin_kernel(
    const float* __restrict__ x_in,        // [B, EMB, CODES]
    const float* __restrict__ codebook,    // [NTOK, EMB]
    float* __restrict__ idx_out)           // [B*CODES] (float-encoded indices)
{
    extern __shared__ char smem[];
    _Float16* xT  = (_Float16*)(smem + X_OFF);   // [QTILE][ROWP]
    _Float16* cT  = (_Float16*)(smem + C_OFF);   // [TCHUNK][ROWP]
    float*    e2s = (float*)   (smem + E_OFF);   // [TCHUNK] : BIAS - 0.5*e2

    const int tid  = threadIdx.x;
    const int lane = tid & 31;
    const int wave = tid >> 5;     // 8 waves
    const int half = lane >> 4;    // 0 | 1
    const int l16  = lane & 15;

    const int blocksPerB = CODES / QTILE;             // 128
    const int b  = blockIdx.x / blocksPerB;
    const int c0 = (blockIdx.x % blocksPerB) * QTILE;

    // ---- stage query tile: transpose [e][c] -> [c][e], f32 -> f16 ----
    {
        const float* xb = x_in + (size_t)b * EMB * CODES + c0;
        for (int i = tid; i < EMB * QTILE; i += 256) {
            int e = i >> 7;      // / 128
            int j = i & 127;
            xT[j * ROWP + e] = (_Float16)xb[(size_t)e * CODES + j];
        }
    }

    // packed (score | 1023-n) running max; score > 0 so 0 is the identity
    unsigned int best[8];
#pragma unroll
    for (int i = 0; i < 8; ++i) best[i] = 0u;

    v16h a0, a1;

    for (int chunk = 0; chunk < NTOK / TCHUNK; ++chunk) {
        const int tt0 = chunk * TCHUNK;

        // ---- stage codebook chunk f32 -> f16 ----
        for (int i = tid; i < TCHUNK * EMB; i += 256) {
            int t = i >> 6;      // / 64
            int e = i & 63;
            cT[t * ROWP + e] = (_Float16)codebook[(size_t)(tt0 + t) * EMB + e];
        }
        // ---- bias terms: BIAS - 0.5 * |e_n|^2 ----
        for (int t = tid; t < TCHUNK; t += 256) {
            const float4* row = (const float4*)(codebook + (size_t)(tt0 + t) * EMB);
            float s = 0.f;
#pragma unroll
            for (int k = 0; k < 16; ++k) {
                float4 v = row[k];
                s += v.x * v.x + v.y * v.y + v.z * v.z + v.w * v.w;
            }
            e2s[t] = BIAS - 0.5f * s;
        }
        __syncthreads();

        if (chunk == 0) {
            // A fragments (16x32 f16 layout): lane<16 row M=l16 K{kb..kb+7, kb+16..kb+23}
            const int row = wave * 16 + l16;
            const int kb  = half * 8;
            const _Float16* base = xT + row * ROWP;
            h8 lo0 = *(const h8*)(base + kb);
            h8 hi0 = *(const h8*)(base + kb + 16);
            h8 lo1 = *(const h8*)(base + 32 + kb);
            h8 hi1 = *(const h8*)(base + 32 + kb + 16);
#pragma unroll
            for (int i = 0; i < 8; ++i) {
                a0[i] = lo0[i]; a0[8 + i] = hi0[i];
                a1[i] = lo1[i]; a1[8 + i] = hi1[i];
            }
        }

        // ---- sweep token tiles, NCHAIN independent accumulator chains per iter ----
        // C operand of the first WMMA is the inline constant 0 (no live-in regs),
        // bias is applied post-WMMA inside the WMMA->VALU hazard shadow.
        const v8f zero = {};
        for (int t = 0; t < TCHUNK / 16; t += NCHAIN) {
            v16h bf0[NCHAIN], bf1[NCHAIN];
            v8f  acc[NCHAIN];
            float        ci[NCHAIN];
            unsigned int inv[NCHAIN];

#pragma unroll
            for (int c = 0; c < NCHAIN; ++c) {
                const int tok = (t + c) * 16 + l16;   // this lane's token column
                inv[c] = (unsigned int)(1023 - (tt0 + tok));
                ci[c]  = e2s[tok];
                // B fragment: col N=l16, K = {half*8..+7, 16+half*8..+7} (frag0), +32 halves (frag1)
                const _Float16* p = cT + tok * ROWP + half * 8;
                bf0[c] = load_bfrag(p);
                bf1[c] = load_bfrag(p + 32);
            }

#pragma unroll
            for (int c = 0; c < NCHAIN; ++c)
                acc[c] = __builtin_amdgcn_wmma_f32_16x16x32_f16(
                             false, a0, false, bf0[c], (short)0, zero, false, false);
#pragma unroll
            for (int c = 0; c < NCHAIN; ++c)
                acc[c] = __builtin_amdgcn_wmma_f32_16x16x32_f16(
                             false, a1, false, bf1[c], (short)0, acc[c], false, false);

            // score = dot + BIAS - 0.5*e2 (> 0); pack score into high 22 bits,
            // (1023-n) into low 10; unsigned max == argmax, ties -> smaller n
#pragma unroll
            for (int c = 0; c < NCHAIN; ++c) {
#pragma unroll
                for (int i = 0; i < 8; ++i) {
                    unsigned int u =
                        (__float_as_uint(acc[c][i] + ci[c]) & 0xFFFFFC00u) | inv[c];
                    best[i] = best[i] > u ? best[i] : u;
                }
            }
        }
        __syncthreads();   // before restaging codebook chunk
    }

    // ---- butterfly max across the 16 lanes of each half ----
#pragma unroll
    for (int i = 0; i < 8; ++i) {
        unsigned int u = best[i];
#pragma unroll
        for (int m = 1; m < 16; m <<= 1) {
            unsigned int o = (unsigned int)__shfl_xor((int)u, m, 32);
            u = u > o ? u : o;
        }
        if (l16 == 0) {
            // C layout: VGPR i -> M=i (lanes 0-15), M=i+8 (lanes 16-31)
            int n = 1023 - (int)(u & 0x3FFu);
            int q = blockIdx.x * QTILE + wave * 16 + half * 8 + i;
            idx_out[q] = (float)n;
        }
    }
}

__global__ __launch_bounds__(256) void vq_gather_kernel(
    const float* __restrict__ codebook,
    const float* __restrict__ idx_f,
    float* __restrict__ x_out)             // [B, EMB, CODES]
{
    int q = blockIdx.x * 256 + threadIdx.x;          // < B*CODES
    int b = q >> 14;
    int c = q & (CODES - 1);
    int idx = (int)idx_f[q];
    const float4* row = (const float4*)(codebook + (size_t)idx * EMB);
    float* out = x_out + (size_t)b * EMB * CODES + c;
#pragma unroll
    for (int k = 0; k < 16; ++k) {
        float4 v = row[k];
        out[(size_t)(4 * k + 0) * CODES] = v.x;
        out[(size_t)(4 * k + 1) * CODES] = v.y;
        out[(size_t)(4 * k + 2) * CODES] = v.z;
        out[(size_t)(4 * k + 3) * CODES] = v.w;
    }
}

extern "C" void kernel_launch(void* const* d_in, const int* in_sizes, int n_in,
                              void* d_out, int out_size, void* d_ws, size_t ws_size,
                              hipStream_t stream) {
    const float* x_in     = (const float*)d_in[0];   // [16, 64, 16384]
    const float* codebook = (const float*)d_in[1];   // [1024, 64]

    float* x_out   = (float*)d_out;                               // 16*64*16384 floats
    float* idx_out = (float*)d_out + (size_t)BATCH * EMB * CODES; // 16*16384 indices

    dim3 grid1(BATCH * CODES / QTILE);   // 2048 blocks
    vq_argmin_kernel<<<grid1, 256, SMEM_BYTES, stream>>>(x_in, codebook, idx_out);

    dim3 grid2(BATCH * CODES / 256);     // 1024 blocks
    vq_gather_kernel<<<grid2, 256, 0, stream>>>(codebook, idx_out, x_out);
}
